// GraphTransformer_73950746902728
// MI455X (gfx1250) — matrix operand order repaired and checked by reading, not compiled, not run on hardware
//
#include <hip/hip_runtime.h>

// ---------------- problem constants ----------------
#define DM     256
#define NHEAD  8
#define NN     2047
#define NB     64
#define NROWS  (NB*NN)          // 131008 (multiple of 16)
#define NVOCAB 32000
#define NTSTEP 5
#define NDEPTH 10

// LDS pitches (padded so 16-lane tile accesses hit distinct banks)
#define APITCH 264              // bf16 elems per A-tile row (256+8)
#define OPITCH 260              // f32  elems per O-tile row (256+4)

typedef __bf16 bf16;
typedef __attribute__((ext_vector_type(16))) __bf16 bf16x16;
typedef __attribute__((ext_vector_type(8)))  __bf16 bf16x8;
typedef __attribute__((ext_vector_type(8)))  float  f32x8;

__device__ __forceinline__ float wave_sum(float p){
#pragma unroll
  for (int off = 16; off > 0; off >>= 1) p += __shfl_xor(p, off, 32);
  return p;
}

// ---- WMMA fragment helpers (wave32, 16x16x32 bf16, f32 accum) ----
// A (16x32 bf16): lane L holds row m=L%16; K = 16h..16h+15 (contiguous per lane half), h=L/16.
__device__ __forceinline__ bf16x16 load_afrag(const bf16* Abuf, int lane, int kstep){
  int m  = lane & 15;
  int k0 = kstep * 32 + ((lane >> 4) << 4);
  const bf16* p = Abuf + m * APITCH + k0;
  bf16x8 lo = *(const bf16x8*)p;
  bf16x8 hi = *(const bf16x8*)(p + 8);
  bf16x16 a;
#pragma unroll
  for (int i = 0; i < 8; i++){ a[i] = lo[i]; a[i+8] = hi[i]; }
  return a;
}
// B (32x16 bf16) from pre-transposed bf16 weights Wt[n*256+k]:
// lane L holds col n=L%16, K = kstep*32 + 16*(L/16) .. +15 (contiguous per lane half).
__device__ __forceinline__ bf16x16 load_bfrag(const bf16* Wt, int lane, int ntile, int kstep){
  int n = (ntile << 4) + (lane & 15);
  int k = kstep * 32 + ((lane >> 4) << 4);
  return *(const bf16x16*)(Wt + (size_t)n * 256 + k);
}

// Fused GEMM: one shared A-fragment per K-step feeds NP weights x 2 N-tiles.
// Cuts LDS A-traffic NPx2-fold vs naive per-tile loops (WMMA wants 2KB/op fed).
template<int NP>
__device__ __forceinline__ void gemm_multi(const bf16* Abuf, int lane, int wave,
                                           const bf16* const* wts, f32x8 acc[NP][2]){
#pragma unroll
  for (int ks = 0; ks < 8; ks++){
    bf16x16 a = load_afrag(Abuf, lane, ks);
#pragma unroll
    for (int p = 0; p < NP; p++){
#pragma unroll
      for (int u = 0; u < 2; u++){
        bf16x16 b = load_bfrag(wts[p], lane, wave * 2 + u, ks);
        acc[p][u] = __builtin_amdgcn_wmma_f32_16x16x32_bf16(
            false, a, false, b, (short)0, acc[p][u], false, false);
      }
    }
  }
}

// D tile (16x16 f32): lane L -> col n=L%16, rows m = 8*(L/16)+v
__device__ __forceinline__ void store_tile_global(float* dst, const f32x8& c, int lane, int ntile,
                                                  const float* bias, const int* rowIdx){
  int n = (ntile << 4) + (lane & 15);
  int mb = (lane >> 4) << 3;
  float bv = bias[n];
#pragma unroll
  for (int v = 0; v < 8; v++)
    dst[(size_t)rowIdx[mb + v] * DM + n] = c[v] + bv;
}

__device__ __forceinline__ void store_tile_lds_relu(float* Obuf, const f32x8& c, int lane, int ntile,
                                                    const float* bias){
  int n = (ntile << 4) + (lane & 15);
  int mb = (lane >> 4) << 3;
  float bv = bias[n];
#pragma unroll
  for (int v = 0; v < 8; v++){
    float x = c[v] + bv;
    Obuf[(mb + v) * OPITCH + n] = x > 0.f ? x : 0.f;
  }
}

// ---------------- weight transpose + f32->bf16 ----------------
__global__ void transpose_w(const float* __restrict__ W, bf16* __restrict__ Wt){
  int i = blockIdx.x * blockDim.x + threadIdx.x;   // 65536
  int n = i >> 8, k = i & 255;
  Wt[(size_t)n * 256 + k] = (bf16)W[(size_t)k * 256 + n];
}

// ---------------- generic C = A(Mx256) @ Wt + bias ----------------
__global__ __launch_bounds__(256) void gemm256(const float* __restrict__ A,
                                               const bf16* __restrict__ Wt,
                                               const float* __restrict__ bias,
                                               float* __restrict__ C){
  __shared__ __attribute__((aligned(32))) bf16 Abuf[16 * APITCH];
  int tid = threadIdx.x, lane = tid & 31, wave = tid >> 5;
  int row0 = blockIdx.x << 4;
  for (int i = tid; i < 16 * 256; i += 256){
    int r = i >> 8, cc = i & 255;
    Abuf[r * APITCH + cc] = (bf16)A[(size_t)(row0 + r) * 256 + cc];
  }
  __syncthreads();
  f32x8 z = {0.f,0.f,0.f,0.f,0.f,0.f,0.f,0.f};
  f32x8 acc[1][2]; acc[0][0] = z; acc[0][1] = z;
  const bf16* w[1] = { Wt };
  gemm_multi<1>(Abuf, lane, wave, w, acc);
#pragma unroll
  for (int u = 0; u < 2; u++){
    int nt = wave * 2 + u;
    int n = (nt << 4) + (lane & 15);
    int mb = (lane >> 4) << 3;
    float bv = bias[n];
#pragma unroll
    for (int v = 0; v < 8; v++)
      C[(size_t)(row0 + mb + v) * 256 + n] = acc[0][u][v] + bv;
  }
}

// ---------------- init Qh/Kh/Vh from vocab tables (h0 = e) ----------------
__global__ void init_qkv(const int* __restrict__ wordid, const int* __restrict__ maskp,
                         const float* __restrict__ Qt, const float* __restrict__ Kt,
                         const float* __restrict__ Vt, const float* __restrict__ mha1_b,
                         float* __restrict__ Qh, float* __restrict__ Kh, float* __restrict__ Vh){
  int grow = blockIdx.x;       // 0..NROWS-1
  int c = threadIdx.x;         // 0..255
  int id = wordid[grow], mm = maskp[grow];
  size_t o = (size_t)grow * 256 + c;
  size_t g = (size_t)id * 256 + c;
  Qh[o] = mm ? Qt[g] : mha1_b[c];
  Kh[o] = mm ? Kt[g] : mha1_b[256 + c];
  Vh[o] = mm ? Vt[g] : mha1_b[512 + c];
}

// ---------------- S = mean_n e[b,n] (two-stage, deterministic) ----------------
__global__ void s_mean_part(const int* __restrict__ wordid, const int* __restrict__ maskp,
                            const float* __restrict__ emb, float* __restrict__ Spart){
  int b = blockIdx.x, chunk = blockIdx.y, c = threadIdx.x;
  int n0 = chunk * 128;
  int n1 = n0 + 128; if (n1 > NN) n1 = NN;
  float acc = 0.f;
  for (int n = n0; n < n1; n++){
    int grow = b * NN + n;
    if (maskp[grow]) acc += emb[(size_t)wordid[grow] * 256 + c];
  }
  Spart[((size_t)chunk * NB + b) * 256 + c] = acc;
}
__global__ void s_mean_red(const float* __restrict__ Spart, float* __restrict__ S){
  int b = blockIdx.x, c = threadIdx.x;
  float acc = 0.f;
  for (int k = 0; k < 16; k++) acc += Spart[((size_t)k * NB + b) * 256 + c];
  S[b * 256 + c] = acc / (float)NN;
}

// ---------------- fused per-level node update ----------------
__global__ __launch_bounds__(256) void level_kernel(
    const int* __restrict__ wordid, const int* __restrict__ maskp,
    float* __restrict__ Qh, float* __restrict__ Kh, float* __restrict__ Vh,
    float* __restrict__ K2h, float* __restrict__ V2h, float* __restrict__ hbuf,
    const float* __restrict__ Kt1, const float* __restrict__ Vt1,
    const float* __restrict__ KS, const float* __restrict__ VS,
    const float* __restrict__ mha1_b, const float* __restrict__ mha2_b,
    const float* __restrict__ ln1_g, const float* __restrict__ ln1_b,
    const bf16* __restrict__ Wt10, const bf16* __restrict__ Wt11,
    const bf16* __restrict__ Wt12, const bf16* __restrict__ Wt13,
    const bf16* __restrict__ Wt21, const bf16* __restrict__ Wt22,
    int lo, int Mn, int isLeaf){
  __shared__ __attribute__((aligned(32))) bf16  Abuf[16 * APITCH];
  __shared__ __attribute__((aligned(16))) float Obuf[16 * OPITCH];
  __shared__ int rowIdx[16];

  int tid = threadIdx.x, lane = tid & 31, wave = tid >> 5;
  int tile0 = blockIdx.x << 4;

  // warm weight matrices into cache (global_prefetch_b8), 256 thr * 512B = 128KB each
  __builtin_prefetch(Wt13 + (size_t)tid * 256, 0, 3);
  __builtin_prefetch(Wt10 + (size_t)tid * 256, 0, 3);
  __builtin_prefetch(Wt11 + (size_t)tid * 256, 0, 3);
  __builtin_prefetch(Wt12 + (size_t)tid * 256, 0, 3);
  __builtin_prefetch(Wt21 + (size_t)tid * 256, 0, 3);
  __builtin_prefetch(Wt22 + (size_t)tid * 256, 0, 3);

  const float* b1 = mha1_b + DM;        // mha1 K bias
  const float* b2 = mha1_b + 2 * DM;    // mha1 V bias
  const float  scale = 0.17677669529663687f;   // 1/sqrt(32)

  // ---- Phase 1: 1x5 attention per item; wave handles items 2w, 2w+1 ----
  for (int s = 0; s < 2; s++){
    int t = wave * 2 + s;
    int item = tile0 + t;
    int b = item / Mn, m = item - b * Mn;
    int idx = lo + m;
    int grow = b * NN + idx;
    if (lane == 0) rowIdx[t] = grow;
    int id = wordid[grow], mm = maskp[grow];

    const float *kp[5], *vp[5];
    if (isLeaf){ kp[0] = b1; vp[0] = b2; kp[1] = b1; vp[1] = b2; }
    else {
      kp[0] = Kh + (size_t)(b * NN + 2 * idx + 1) * DM;
      vp[0] = Vh + (size_t)(b * NN + 2 * idx + 1) * DM;
      kp[1] = Kh + (size_t)(b * NN + 2 * idx + 2) * DM;
      vp[1] = Vh + (size_t)(b * NN + 2 * idx + 2) * DM;
    }
    kp[2] = Kh + (size_t)grow * DM;  vp[2] = Vh + (size_t)grow * DM;
    kp[3] = mm ? (Kt1 + (size_t)id * DM) : b1;
    vp[3] = mm ? (Vt1 + (size_t)id * DM) : b2;
    kp[4] = KS + b * DM;             vp[4] = VS + b * DM;
    const float* qp = Qh + (size_t)grow * DM;

    for (int hd = 0; hd < NHEAD; hd++){
      int c = hd * 32 + lane;
      float q = qp[c];
      float sc[5];
#pragma unroll
      for (int j = 0; j < 5; j++) sc[j] = wave_sum(q * kp[j][c]) * scale;
      float mx = sc[0];
#pragma unroll
      for (int j = 1; j < 5; j++) mx = fmaxf(mx, sc[j]);
      float a[5], den = 0.f;
#pragma unroll
      for (int j = 0; j < 5; j++){ a[j] = expf(sc[j] - mx); den += a[j]; }
      float inv = 1.f / den;
      float o = 0.f;
#pragma unroll
      for (int j = 0; j < 5; j++) o += a[j] * inv * vp[j][c];
      Abuf[t * APITCH + c] = (bf16)o;
    }
  }
  __syncthreads();

  // ---- Phase 2: out-projection (o @ W3 + b3), relu -> Obuf ----
  {
    const float* b3 = mha1_b + 3 * DM;
    f32x8 z = {0.f,0.f,0.f,0.f,0.f,0.f,0.f,0.f};
    f32x8 acc[1][2]; acc[0][0] = z; acc[0][1] = z;
    const bf16* w[1] = { Wt13 };
    gemm_multi<1>(Abuf, lane, wave, w, acc);
#pragma unroll
    for (int u = 0; u < 2; u++)
      store_tile_lds_relu(Obuf, acc[0][u], lane, wave * 2 + u, b3);
  }
  __syncthreads();

  // ---- Phase 3: layernorm per row -> hbuf (f32) and Abuf (bf16) ----
  for (int s = 0; s < 2; s++){
    int t = wave * 2 + s;
    int grow = rowIdx[t];
    const float* orow = Obuf + t * OPITCH;
    float sum = 0.f;
    for (int c = lane; c < 256; c += 32) sum += orow[c];
    float mu = wave_sum(sum) * (1.f / 256.f);
    float ss = 0.f;
    for (int c = lane; c < 256; c += 32){ float d = orow[c] - mu; ss += d * d; }
    ss = wave_sum(ss);
    float rstd = 1.f / (sqrtf(ss * (1.f / 255.f)) + 1e-6f);
    for (int c = lane; c < 256; c += 32){
      float y = ln1_g[c] * (orow[c] - mu) * rstd + ln1_b[c];
      hbuf[(size_t)grow * DM + c] = y;
      Abuf[t * APITCH + c] = (bf16)y;
    }
  }
  __syncthreads();

  // ---- Phase 4: 5 cached projections of hnew, single A-fragment feed ----
  {
    const float* biases[5] = { mha1_b, mha1_b + DM, mha1_b + 2*DM, mha2_b + DM, mha2_b + 2*DM };
    float*       dsts[5]   = { Qh, Kh, Vh, K2h, V2h };
    const bf16*  wts[5]    = { Wt10, Wt11, Wt12, Wt21, Wt22 };
    f32x8 z = {0.f,0.f,0.f,0.f,0.f,0.f,0.f,0.f};
    f32x8 acc[5][2];
#pragma unroll
    for (int p = 0; p < 5; p++){ acc[p][0] = z; acc[p][1] = z; }
    gemm_multi<5>(Abuf, lane, wave, wts, acc);
#pragma unroll
    for (int p = 0; p < 5; p++)
#pragma unroll
      for (int u = 0; u < 2; u++)
        store_tile_global(dsts[p], acc[p][u], lane, wave * 2 + u, biases[p], rowIdx);
  }
}

// ---------------- mha2: S attends to all N h-rows (S-row masked out) ----------------
__global__ __launch_bounds__(256) void s_attention(const float* __restrict__ Q2,
                                                   const float* __restrict__ K2h,
                                                   const float* __restrict__ V2h,
                                                   float* __restrict__ o2){
  __shared__ float sc[NN];
  __shared__ float red[8];
  __shared__ float ov[8][32];
  int b = blockIdx.x >> 3, hd = blockIdx.x & 7;
  int lane = threadIdx.x & 31, wave = threadIdx.x >> 5;
  const float scale = 0.17677669529663687f;
  float q = Q2[b * 256 + hd * 32 + lane];
  float mloc = -1e30f;
  for (int n = wave; n < NN; n += 8){
    float p = wave_sum(q * K2h[((size_t)(b * NN + n)) * 256 + hd * 32 + lane]) * scale;
    if (lane == 0) sc[n] = p;
    mloc = fmaxf(mloc, p);
  }
  if (lane == 0) red[wave] = mloc;
  __syncthreads();
  float mx = red[0];
#pragma unroll
  for (int i = 1; i < 8; i++) mx = fmaxf(mx, red[i]);
  float sloc = 0.f, oacc = 0.f;
  for (int n = wave; n < NN; n += 8){
    float a = expf(sc[n] - mx);
    sloc += a;
    oacc += a * V2h[((size_t)(b * NN + n)) * 256 + hd * 32 + lane];
  }
  __syncthreads();
  if (lane == 0) red[wave] = sloc;
  ov[wave][lane] = oacc;
  __syncthreads();
  if (wave == 0){
    float tot = 0.f;
#pragma unroll
    for (int i = 0; i < 8; i++) tot += red[i];
    float o = 0.f;
#pragma unroll
    for (int i = 0; i < 8; i++) o += ov[i][lane];
    o2[b * 256 + hd * 32 + lane] = o / tot;
  }
}

// ---------------- relu + layernorm on 64x256 (S update) ----------------
__global__ void relu_ln_kernel(const float* __restrict__ X, const float* __restrict__ g,
                               const float* __restrict__ bb, float* __restrict__ Y){
  int row = blockIdx.x, lane = threadIdx.x;   // block = 32
  const float* x = X + (size_t)row * 256;
  float v[8], sum = 0.f;
#pragma unroll
  for (int i = 0; i < 8; i++){ float t = x[lane + 32 * i]; t = t > 0.f ? t : 0.f; v[i] = t; sum += t; }
  float mu = wave_sum(sum) * (1.f / 256.f);
  float ss = 0.f;
#pragma unroll
  for (int i = 0; i < 8; i++){ float d = v[i] - mu; ss += d * d; }
  ss = wave_sum(ss);
  float rstd = 1.f / (sqrtf(ss * (1.f / 255.f)) + 1e-6f);
#pragma unroll
  for (int i = 0; i < 8; i++){
    int c = lane + 32 * i;
    Y[(size_t)row * 256 + c] = g[c] * (v[i] - mu) * rstd + bb[c];
  }
}

// ---------------- logits: out = h @ out_W + out_b ----------------
__global__ __launch_bounds__(256) void logits_kernel(const float* __restrict__ h,
                                                     const float* __restrict__ W,
                                                     const float* __restrict__ bias,
                                                     float* __restrict__ out){
  int lane = threadIdx.x & 31, wave = threadIdx.x >> 5;
  size_t row = (size_t)blockIdx.x * 8 + wave;
  const float* hr = h + row * 256;
  float a0 = 0.f, a1 = 0.f, a2 = 0.f, a3 = 0.f;
  for (int c = lane; c < 256; c += 32){
    float hv = hr[c];
    a0 += hv * W[c * 4 + 0]; a1 += hv * W[c * 4 + 1];
    a2 += hv * W[c * 4 + 2]; a3 += hv * W[c * 4 + 3];
  }
  a0 = wave_sum(a0); a1 = wave_sum(a1); a2 = wave_sum(a2); a3 = wave_sum(a3);
  if (lane == 0){
    out[row * 4 + 0] = a0 + bias[0];
    out[row * 4 + 1] = a1 + bias[1];
    out[row * 4 + 2] = a2 + bias[2];
    out[row * 4 + 3] = a3 + bias[3];
  }
}

// ---------------- host orchestration ----------------
extern "C" void kernel_launch(void* const* d_in, const int* in_sizes, int n_in,
                              void* d_out, int out_size, void* d_ws, size_t ws_size,
                              hipStream_t stream){
  (void)in_sizes; (void)n_in; (void)out_size; (void)ws_size;
  const int*   wordid = (const int*)d_in[0];
  const int*   maskp  = (const int*)d_in[1];
  const float* emb    = (const float*)d_in[2];
  const float* mha1_W = (const float*)d_in[3];
  const float* mha1_b = (const float*)d_in[4];
  const float* mha2_W = (const float*)d_in[5];
  const float* mha2_b = (const float*)d_in[6];
  const float* ln1_g  = (const float*)d_in[7];
  const float* ln1_b  = (const float*)d_in[8];
  const float* ln2_g  = (const float*)d_in[9];
  const float* ln2_b  = (const float*)d_in[10];
  const float* out_W  = (const float*)d_in[11];
  const float* out_b  = (const float*)d_in[12];
  float* out = (float*)d_out;

  char* ws = (char*)d_ws;
  size_t off = 0;
  auto alloc = [&](size_t bytes)->char*{
    char* p = ws + off; off += (bytes + 255) & ~(size_t)255; return p;
  };
  const size_t big = (size_t)NROWS * 256 * sizeof(float);       // 134.2 MB
  float* Qh   = (float*)alloc(big);
  float* Kh   = (float*)alloc(big);
  float* Vh   = (float*)alloc(big);
  float* K2h  = (float*)alloc(big);
  float* V2h  = (float*)alloc(big);
  float* hbuf = (float*)alloc(big);
  const size_t tbl = (size_t)NVOCAB * 256 * sizeof(float);      // 32.8 MB
  float* Qt1 = (float*)alloc(tbl);
  float* Kt1 = (float*)alloc(tbl);
  float* Vt1 = (float*)alloc(tbl);
  bf16* Wt[8];
  for (int i = 0; i < 8; i++) Wt[i] = (bf16*)alloc((size_t)256 * 256 * sizeof(bf16));
  float* S     = (float*)alloc((size_t)NB * 256 * sizeof(float));
  float* Spart = (float*)alloc((size_t)16 * NB * 256 * sizeof(float));
  float* KS    = (float*)alloc((size_t)NB * 256 * sizeof(float));
  float* VS    = (float*)alloc((size_t)NB * 256 * sizeof(float));
  float* Q2    = (float*)alloc((size_t)NB * 256 * sizeof(float));
  float* o2    = (float*)alloc((size_t)NB * 256 * sizeof(float));
  float* SO    = (float*)alloc((size_t)NB * 256 * sizeof(float));
  // total ws usage ~906 MB

  // bf16-transposed weights: Wt[0..3] = mha1 W0..W3, Wt[4..7] = mha2 W0..W3
  for (int j = 0; j < 4; j++)
    transpose_w<<<256, 256, 0, stream>>>(mha1_W + (size_t)j * 65536, Wt[j]);
  for (int j = 0; j < 4; j++)
    transpose_w<<<256, 256, 0, stream>>>(mha2_W + (size_t)j * 65536, Wt[4 + j]);

  // vocab projection tables (bias folded in)
  gemm256<<<NVOCAB / 16, 256, 0, stream>>>(emb, Wt[0], mha1_b,       Qt1);
  gemm256<<<NVOCAB / 16, 256, 0, stream>>>(emb, Wt[1], mha1_b + 256, Kt1);
  gemm256<<<NVOCAB / 16, 256, 0, stream>>>(emb, Wt[2], mha1_b + 512, Vt1);

  // h0 = e  ->  init cached mha1 projections
  init_qkv<<<NROWS, 256, 0, stream>>>(wordid, maskp, Qt1, Kt1, Vt1, mha1_b, Qh, Kh, Vh);

  // S0 = mean_n e[b,n]
  s_mean_part<<<dim3(NB, 16), 256, 0, stream>>>(wordid, maskp, emb, Spart);
  s_mean_red<<<NB, 256, 0, stream>>>(Spart, S);

  for (int t = 0; t < NTSTEP; t++){
    // S-dependent projections (same S used for mha1 ctx row and mha2 query this timestep)
    gemm256<<<NB / 16, 256, 0, stream>>>(S, Wt[1], mha1_b + 256, KS);
    gemm256<<<NB / 16, 256, 0, stream>>>(S, Wt[2], mha1_b + 512, VS);
    gemm256<<<NB / 16, 256, 0, stream>>>(S, Wt[4], mha2_b,       Q2);

    for (int l = NDEPTH; l >= 0; l--){
      int Mn = 1 << l, lo = Mn - 1;
      int nblk = (NB * Mn) / 16;       // always exact (NB=64)
      level_kernel<<<nblk, 256, 0, stream>>>(wordid, maskp,
          Qh, Kh, Vh, K2h, V2h, hbuf, Kt1, Vt1, KS, VS,
          mha1_b, mha2_b, ln1_g, ln1_b,
          Wt[0], Wt[1], Wt[2], Wt[3], Wt[5], Wt[6],
          lo, Mn, (l == NDEPTH) ? 1 : 0);
    }

    s_attention<<<NB * NHEAD, 256, 0, stream>>>(Q2, K2h, V2h, o2);
    gemm256<<<NB / 16, 256, 0, stream>>>(o2, Wt[7], mha2_b + 3 * 256, SO);
    relu_ln_kernel<<<NB, 32, 0, stream>>>(SO, ln2_g, ln2_b, S);
  }

  logits_kernel<<<NROWS / 8, 256, 0, stream>>>(hbuf, out_W, out_b, out);
}
